// PtEncoder_46248207843886
// MI455X (gfx1250) — matrix operand order repaired and checked by reading, not compiled, not run on hardware
//
#include <hip/hip_runtime.h>
#include <hip/hip_bf16.h>
#include <cstdint>

typedef __attribute__((ext_vector_type(16))) _Float16 v16h;
typedef __attribute__((ext_vector_type(8)))  float    v8f;

// ---------------------------------------------------------------------------
// WMMA helpers (CDNA5 gfx1250, wave32, V_WMMA_F32_16X16X32_F16)
// A fragment: 16x32 f16, lane m = lane&15, kbase = (lane>>4)*16,
//             element e -> k = kbase + e   (one contiguous 32B run)
// B fragment: 32x16 f16, lane n = lane&15, kbase = (lane>>4)*16, k = kbase+e
//             (one contiguous 32B run -> two global_load_b128)
// C/D       : 16x16 f32, lane n = lane&15, row  = vgpr_i + (lane>=16 ? 8 : 0)
// ---------------------------------------------------------------------------

__device__ __forceinline__ v8f zero8() {
  v8f z;
#pragma unroll
  for (int i = 0; i < 8; ++i) z[i] = 0.0f;
  return z;
}

__device__ __forceinline__ v16h load_a16(const _Float16* A, int ldk, int k0, int lane) {
  int m  = lane & 15;
  int kb = k0 + ((lane >> 4) << 4);
  const _Float16* p = A + (size_t)m * ldk + kb;
  v16h a;
#pragma unroll
  for (int e = 0; e < 8; ++e) a[e] = p[e];        // k = kb .. kb+7
#pragma unroll
  for (int e = 0; e < 8; ++e) a[8 + e] = p[8 + e]; // k = kb+8 .. kb+15
  return a;
}

// Weights pre-converted to f16, rows zero-padded to ldw (multiple of 32):
// branch-free contiguous 32B per lane.
__device__ __forceinline__ v16h load_b16(const _Float16* W, int ldw, int n0, int k0, int lane) {
  int nn = n0 + (lane & 15);
  int kb = k0 + ((lane >> 4) << 4);
  const _Float16* p = W + (size_t)nn * ldw + kb;
  v16h b;
#pragma unroll
  for (int e = 0; e < 16; ++e) b[e] = p[e];
  return b;
}

template <int MT>
__device__ __forceinline__ void gemm_tiles(const _Float16* A, int ldk, int kin,
                                           const _Float16* W, int n0,
                                           int lane, v8f acc[MT]) {
  for (int k0 = 0; k0 < kin; k0 += 32) {
    v16h bf = load_b16(W, kin, n0, k0, lane);
#pragma unroll
    for (int t = 0; t < MT; ++t) {
      v16h af = load_a16(A + (size_t)t * 16 * ldk, ldk, k0, lane);
      acc[t] = __builtin_amdgcn_wmma_f32_16x16x32_f16(
          false, af, false, bf, (short)0, acc[t], false, false);
    }
  }
}

// One MLP layer: Ain (LDS f16, ld=ldk, valid K = kin) x W^T + b, ReLU,
// write f16 to Aout (ld=ldko), zero-pad columns [cout, kout_pad).
template <int MT>
__device__ __forceinline__ void sa_layer(const _Float16* Ain, int ldk, int kin,
                                         const _Float16* W, const float* bias,
                                         int cout, int kout_pad,
                                         _Float16* Aout, int ldko, int lane) {
  for (int s = 0; s < MT * 16; ++s)
    for (int c = cout + lane; c < kout_pad; c += 32)
      Aout[(size_t)s * ldko + c] = (_Float16)0.0f;

  for (int n0 = 0; n0 < cout; n0 += 16) {
    v8f acc[MT];
#pragma unroll
    for (int t = 0; t < MT; ++t) acc[t] = zero8();
    gemm_tiles<MT>(Ain, ldk, kin, W, n0, lane, acc);
    float bn = bias[n0 + (lane & 15)];
    int   nn = n0 + (lane & 15);
    int   hb = (lane >> 4) << 3;
#pragma unroll
    for (int t = 0; t < MT; ++t)
#pragma unroll
      for (int i = 0; i < 8; ++i) {
        float v = fmaxf(acc[t][i] + bn, 0.0f);
        Aout[(size_t)(t * 16 + hb + i) * ldko + nn] = (_Float16)v;
      }
  }
}

// Final layer + max-pool over all MT*16 rows, f32 output (cout channels).
template <int MT>
__device__ __forceinline__ void sa_layer_max(const _Float16* Ain, int ldk, int kin,
                                             const _Float16* W, const float* bias,
                                             int cout, int lane, float* out) {
  for (int n0 = 0; n0 < cout; n0 += 16) {
    v8f acc[MT];
#pragma unroll
    for (int t = 0; t < MT; ++t) acc[t] = zero8();
    gemm_tiles<MT>(Ain, ldk, kin, W, n0, lane, acc);
    float bn = bias[n0 + (lane & 15)];
    float mx = 0.0f;  // ReLU output is >= 0, so 0 is a safe identity
#pragma unroll
    for (int t = 0; t < MT; ++t)
#pragma unroll
      for (int i = 0; i < 8; ++i)
        mx = fmaxf(mx, fmaxf(acc[t][i] + bn, 0.0f));
    mx = fmaxf(mx, __shfl_xor(mx, 16));
    if (lane < 16) out[n0 + lane] = mx;
  }
}

// ---------------------------------------------------------------------------
// Weight pre-conversion: f32 [cout][cin] -> f16 [cout][kpad], zero-padded.
// ---------------------------------------------------------------------------
__global__ void wcvt_kernel(const float* __restrict__ W, _Float16* __restrict__ out,
                            int cin, int kpad, int total) {
  int t = blockIdx.x * blockDim.x + threadIdx.x;
  if (t >= total) return;
  int r = t / kpad, c = t % kpad;
  out[t] = (_Float16)((c < cin) ? W[(size_t)r * cin + c] : 0.0f);
}

// ---------------------------------------------------------------------------
// Farthest point sampling: one block per batch, 1024 threads, per-point
// running min-distance kept in registers, LDS tree-argmax per iteration.
// ---------------------------------------------------------------------------
__global__ __launch_bounds__(1024) void fps_kernel(const float* __restrict__ pts,
                                                   int* __restrict__ idx_out,
                                                   int n, int npoint) {
  __shared__ float sval[1024];
  __shared__ int   sidx[1024];
  int b = blockIdx.x, tid = threadIdx.x;
  const float* P = pts + (size_t)b * n * 3;
  float dist[8];
#pragma unroll
  for (int j = 0; j < 8; ++j) dist[j] = 1e10f;
  int last = 0;
  if (tid == 0) idx_out[(size_t)b * npoint] = 0;

  for (int it = 1; it < npoint; ++it) {
    float lx = P[last * 3 + 0], ly = P[last * 3 + 1], lz = P[last * 3 + 2];
    float bv = -1.0f; int bi = 0;
#pragma unroll
    for (int j = 0; j < 8; ++j) {
      int p = tid + j * 1024;
      if (p < n) {
        float dx = P[p * 3 + 0] - lx, dy = P[p * 3 + 1] - ly, dz = P[p * 3 + 2] - lz;
        float d = fminf(dist[j], dx * dx + dy * dy + dz * dz);
        dist[j] = d;
        if (d > bv) { bv = d; bi = p; }
      }
    }
    sval[tid] = bv; sidx[tid] = bi;
    __syncthreads();
    for (int s = 512; s > 0; s >>= 1) {
      if (tid < s && sval[tid + s] > sval[tid]) {
        sval[tid] = sval[tid + s]; sidx[tid] = sidx[tid + s];
      }
      __syncthreads();
    }
    last = sidx[0];
    if (tid == 0) idx_out[(size_t)b * npoint + it] = last;
    __syncthreads();
  }
}

__global__ void gather_xyz_kernel(const float* __restrict__ pts,
                                  const int* __restrict__ idx,
                                  float* __restrict__ out,
                                  int n, int npoint, int B) {
  int t = blockIdx.x * blockDim.x + threadIdx.x;
  int total = B * npoint * 3;
  if (t >= total) return;
  int c = t % 3;
  int m = (t / 3) % npoint;
  int b = t / (3 * npoint);
  out[t] = pts[((size_t)b * n + idx[(size_t)b * npoint + m]) * 3 + c];
}

// ---------------------------------------------------------------------------
// Ball query: one wave per query point; first-k smallest indices in ball
// (matches the reference's sorted-key selection), pad with first hit (or 0).
// ---------------------------------------------------------------------------
__global__ __launch_bounds__(256) void ball_query_kernel(
    const float* __restrict__ pts, const float* __restrict__ centers,
    int* __restrict__ gidx, int B, int n, int npoint, float r2, int k) {
  int gid  = blockIdx.x * 8 + (threadIdx.x >> 5);
  int lane = threadIdx.x & 31;
  if (gid >= B * npoint) return;
  int b = gid / npoint;
  float cx = centers[(size_t)gid * 3 + 0];
  float cy = centers[(size_t)gid * 3 + 1];
  float cz = centers[(size_t)gid * 3 + 2];
  const float* P = pts + (size_t)b * n * 3;

  int cnt = 0, slotval = 0;
  for (int j0 = 0; j0 < n && cnt < k; j0 += 32) {
    int j = j0 + lane;
    float dx = P[j * 3 + 0] - cx, dy = P[j * 3 + 1] - cy, dz = P[j * 3 + 2] - cz;
    float d2 = dx * dx + dy * dy + dz * dz;
    unsigned mask = (unsigned)__ballot(d2 < r2);
    while (mask && cnt < k) {
      int bit = __ffs(mask) - 1;
      int jv  = __shfl(j, bit);
      if (lane == cnt) slotval = jv;
      ++cnt;
      mask &= mask - 1;
    }
  }
  int pad = __shfl(slotval, 0);
  if (cnt == 0) pad = 0;
  int v = (lane < cnt) ? slotval : pad;
  if (lane < k) gidx[(size_t)gid * k + lane] = v;
}

// ---------------------------------------------------------------------------
// Fused SA kernel: one wave per query point. Group gather -> 3 WMMA MLP
// layers (f16 in / f32 acc) -> channel-wise max over nsample rows.
// ---------------------------------------------------------------------------
template <int NS, int CF, int C1, int C2, int C3>
__global__ __launch_bounds__(64) void sa_kernel(
    const float* __restrict__ pts, const float* __restrict__ feats,
    const float* __restrict__ centers, const int* __restrict__ gidx,
    const _Float16* __restrict__ W0, const float* __restrict__ b0,
    const _Float16* __restrict__ W1, const float* __restrict__ b1,
    const _Float16* __restrict__ W2, const float* __restrict__ b2,
    float* __restrict__ out, int n, int npoint, int ctot, int coff) {
  constexpr int C0 = 3 + CF;
  constexpr int K0 = (C0 + 31) / 32 * 32;
  constexpr int K1 = (C1 + 31) / 32 * 32;
  constexpr int K2 = (C2 + 31) / 32 * 32;
  constexpr int KA = (K0 > K1) ? K0 : K1;
  constexpr int KMAX = (KA > K2) ? KA : K2;
  constexpr int NW = 2;
  constexpr int MT = NS / 16;

  __shared__ _Float16 buf[NW][2][NS][KMAX];

  int w = threadIdx.x >> 5, lane = threadIdx.x & 31;
  int gq = blockIdx.x * NW + w;
  int b = gq / npoint;

  const int* gi = gidx + (size_t)gq * NS;
  float cx = centers[(size_t)gq * 3 + 0];
  float cy = centers[(size_t)gq * 3 + 1];
  float cz = centers[(size_t)gq * 3 + 2];

  _Float16* A0 = &buf[w][0][0][0];
  _Float16* A1 = &buf[w][1][0][0];

  // Stage grouped input (relative xyz ++ gathered feats), zero-padded to K0.
  for (int s = 0; s < NS; ++s) {
    int pi = gi[s];
    const float* pp = pts + ((size_t)b * n + pi) * 3;
    for (int c = lane; c < K0; c += 32) {
      float v = 0.0f;
      if      (c == 0) v = pp[0] - cx;
      else if (c == 1) v = pp[1] - cy;
      else if (c == 2) v = pp[2] - cz;
      else if (CF > 0 && c < 3 + CF) v = feats[((size_t)b * n + pi) * CF + (c - 3)];
      A0[(size_t)s * KMAX + c] = (_Float16)v;
    }
  }

  sa_layer<MT>(A0, KMAX, K0, W0, b0, C1, K1, A1, KMAX, lane);
  sa_layer<MT>(A1, KMAX, K1, W1, b1, C2, K2, A0, KMAX, lane);
  sa_layer_max<MT>(A0, KMAX, K2, W2, b2, C3, lane,
                   out + (size_t)gq * ctot + coff);
}

// ---------------------------------------------------------------------------
// 3-NN interpolation: one wave per unknown point; per-lane top-3 then
// butterfly shfl merge; writes concatenated (unknown_feats ++ interp) rows
// directly as f16 into workspace for the FP MLP.
// ---------------------------------------------------------------------------
__global__ __launch_bounds__(256) void interp_kernel(
    const float* __restrict__ ux, const float* __restrict__ kx,
    const float* __restrict__ uf, const float* __restrict__ kf,
    _Float16* __restrict__ xin, int B, int Np, int Nk, int Cu, int Ck) {
  int gid  = blockIdx.x * 8 + (threadIdx.x >> 5);
  int lane = threadIdx.x & 31;
  if (gid >= B * Np) return;
  int b = gid / Np;

  float qx = ux[(size_t)gid * 3 + 0];
  float qy = ux[(size_t)gid * 3 + 1];
  float qz = ux[(size_t)gid * 3 + 2];

  float d[3] = {1e30f, 1e30f, 1e30f};
  int   ix[3] = {0, 0, 0};
  const float* K = kx + (size_t)b * Nk * 3;
  for (int j = lane; j < Nk; j += 32) {
    float dx = K[j * 3 + 0] - qx, dy = K[j * 3 + 1] - qy, dz = K[j * 3 + 2] - qz;
    float d2 = dx * dx + dy * dy + dz * dz;
    if (d2 < d[2]) {
      if (d2 < d[0])      { d[2]=d[1]; ix[2]=ix[1]; d[1]=d[0]; ix[1]=ix[0]; d[0]=d2; ix[0]=j; }
      else if (d2 < d[1]) { d[2]=d[1]; ix[2]=ix[1]; d[1]=d2; ix[1]=j; }
      else                { d[2]=d2; ix[2]=j; }
    }
  }
  // Butterfly merge of sorted triples across the wave.
  for (int off = 16; off > 0; off >>= 1) {
    float md[6]; int mi[6];
#pragma unroll
    for (int j = 0; j < 3; ++j) {
      md[j]     = d[j];                  mi[j]     = ix[j];
      md[j + 3] = __shfl_xor(d[j], off); mi[j + 3] = __shfl_xor(ix[j], off);
    }
#pragma unroll
    for (int r = 0; r < 3; ++r) {
      int best = r;
      for (int t = r + 1; t < 6; ++t)
        if (md[t] < md[best]) best = t;
      float tv = md[r]; md[r] = md[best]; md[best] = tv;
      int   ti = mi[r]; mi[r] = mi[best]; mi[best] = ti;
      d[r] = md[r]; ix[r] = mi[r];
    }
  }
  float w0 = 1.0f / (d[0] + 1e-8f);
  float w1 = 1.0f / (d[1] + 1e-8f);
  float w2 = 1.0f / (d[2] + 1e-8f);
  float ws = w0 + w1 + w2;
  w0 /= ws; w1 /= ws; w2 /= ws;

  const float* f0 = kf + ((size_t)b * Nk + ix[0]) * Ck;
  const float* f1 = kf + ((size_t)b * Nk + ix[1]) * Ck;
  const float* f2 = kf + ((size_t)b * Nk + ix[2]) * Ck;
  _Float16* xr = xin + (size_t)gid * (Cu + Ck);
  for (int c = lane; c < Cu; c += 32)
    xr[c] = (_Float16)uf[(size_t)gid * Cu + c];
  for (int c = lane; c < Ck; c += 32)
    xr[Cu + c] = (_Float16)(w0 * f0[c] + w1 * f1[c] + w2 * f2[c]);
}

// ---------------------------------------------------------------------------
// FP pointwise MLP (2 layers, WMMA): 16 rows per wave, 4 waves per block.
// Layer-0 input rows come straight from the f16 workspace matrix.
// ---------------------------------------------------------------------------
template <int C0, int C1, int C2>
__global__ __launch_bounds__(128) void fp_mlp_kernel(
    const _Float16* __restrict__ Xin,
    const _Float16* __restrict__ W0, const float* __restrict__ b0,
    const _Float16* __restrict__ W1, const float* __restrict__ b1,
    float* __restrict__ out, int Mtot) {
  constexpr int NW = 4;
  __shared__ _Float16 mid[NW][16][C1];
  int w = threadIdx.x >> 5, lane = threadIdx.x & 31;
  int row0 = (blockIdx.x * NW + w) * 16;
  if (row0 >= Mtot) return;

  const _Float16* A = Xin + (size_t)row0 * C0;
  _Float16* M = &mid[w][0][0];
  int hb = (lane >> 4) << 3;
  int nl = lane & 15;

  for (int n0 = 0; n0 < C1; n0 += 16) {
    v8f acc = zero8();
    for (int k0 = 0; k0 < C0; k0 += 32) {
      v16h bf = load_b16(W0, C0, n0, k0, lane);
      v16h af = load_a16(A, C0, k0, lane);
      acc = __builtin_amdgcn_wmma_f32_16x16x32_f16(
          false, af, false, bf, (short)0, acc, false, false);
    }
    float bn = b0[n0 + nl];
#pragma unroll
    for (int i = 0; i < 8; ++i)
      M[(size_t)(hb + i) * C1 + n0 + nl] = (_Float16)fmaxf(acc[i] + bn, 0.0f);
  }

  for (int n0 = 0; n0 < C2; n0 += 16) {
    v8f acc = zero8();
    for (int k0 = 0; k0 < C1; k0 += 32) {
      v16h bf = load_b16(W1, C1, n0, k0, lane);
      v16h af = load_a16(M, C1, k0, lane);
      acc = __builtin_amdgcn_wmma_f32_16x16x32_f16(
          false, af, false, bf, (short)0, acc, false, false);
    }
    float bn = b1[n0 + nl];
#pragma unroll
    for (int i = 0; i < 8; ++i)
      out[(size_t)(row0 + hb + i) * C2 + n0 + nl] = fmaxf(acc[i] + bn, 0.0f);
  }
}

// ---------------------------------------------------------------------------
// Host-side launch sequence
// ---------------------------------------------------------------------------
extern "C" void kernel_launch(void* const* d_in, const int* in_sizes, int n_in,
                              void* d_out, int out_size, void* d_ws, size_t ws_size,
                              hipStream_t stream) {
  (void)in_sizes; (void)n_in; (void)out_size; (void)ws_size;
  constexpr int B = 8, N = 8192, NP1 = 1024, NP2 = 256;

  const float* xyz = (const float*)d_in[0];
  // d_in layout: [0]=pointcloud, [1..24]=SA (2 lvl x 2 scale x 3 layer x {W,b}),
  //              [25..32]=FP (fp0 W,b,W,b then fp1 W,b,W,b)
  const float* wsrc[16]; const float* bsrc[16];
  {
    int p = 1;
    for (int i = 0; i < 12; ++i) {            // SA: scales 0..3, layers 0..2
      wsrc[i] = (const float*)d_in[p++];
      bsrc[i] = (const float*)d_in[p++];
    }
    for (int i = 12; i < 16; ++i) {           // FP0 L0, FP0 L1, FP1 L0, FP1 L1
      wsrc[i] = (const float*)d_in[p++];
      bsrc[i] = (const float*)d_in[p++];
    }
  }
  // (cin, cout) per matrix, same order as wsrc.
  const int wcin[16]  = {3,16,16,  3,32,32,  99,64,64,  99,64,96,  256,128, 352,256};
  const int wcout[16] = {16,16,32, 32,32,64, 64,64,128, 64,96,128, 128,128, 256,256};

  char* ws = (char*)d_ws;
  size_t o = 0;
  int*       fps_idx1 = (int*)(ws + o);       o += (size_t)B * NP1 * 4;
  float*     new_xyz1 = (float*)(ws + o);     o += (size_t)B * NP1 * 3 * 4;
  int*       fps_idx2 = (int*)(ws + o);       o += (size_t)B * NP2 * 4;
  float*     new_xyz2 = (float*)(ws + o);     o += (size_t)B * NP2 * 3 * 4;
  float*     feats1   = (float*)(ws + o);     o += (size_t)B * NP1 * 96 * 4;
  float*     feats2   = (float*)(ws + o);     o += (size_t)B * NP2 * 256 * 4;
  int*       gidx     = (int*)(ws + o);       o += (size_t)B * NP1 * 32 * 4;
  _Float16*  xin_fp1  = (_Float16*)(ws + o);  o += (size_t)B * NP1 * 352 * 2;
  float*     fp1_out  = (float*)(ws + o);     o += (size_t)B * NP1 * 256 * 4;
  _Float16*  xin_fp0  = (_Float16*)(ws + o);  o += (size_t)B * N * 256 * 2;

  // f16 weight region (zero-padded rows) + conversion launches.
  _Float16* wh[16];
  for (int i = 0; i < 16; ++i) {
    int kpad = (wcin[i] + 31) / 32 * 32;
    int total = wcout[i] * kpad;
    wh[i] = (_Float16*)(ws + o);
    o += (size_t)total * 2;
    wcvt_kernel<<<(total + 255) / 256, 256, 0, stream>>>(wsrc[i], wh[i],
                                                         wcin[i], kpad, total);
  }

  // ---- SA level 0 (8192 -> 1024, feats: none -> 96ch) ----
  fps_kernel<<<B, 1024, 0, stream>>>(xyz, fps_idx1, N, NP1);
  gather_xyz_kernel<<<(B * NP1 * 3 + 255) / 256, 256, 0, stream>>>(
      xyz, fps_idx1, new_xyz1, N, NP1, B);

  ball_query_kernel<<<B * NP1 / 8, 256, 0, stream>>>(
      xyz, new_xyz1, gidx, B, N, NP1, 0.1f * 0.1f, 16);
  sa_kernel<16, 0, 16, 16, 32><<<B * NP1 / 2, 64, 0, stream>>>(
      xyz, nullptr, new_xyz1, gidx,
      wh[0], bsrc[0], wh[1], bsrc[1], wh[2], bsrc[2],
      feats1, N, NP1, 96, 0);

  ball_query_kernel<<<B * NP1 / 8, 256, 0, stream>>>(
      xyz, new_xyz1, gidx, B, N, NP1, 0.5f * 0.5f, 32);
  sa_kernel<32, 0, 32, 32, 64><<<B * NP1 / 2, 64, 0, stream>>>(
      xyz, nullptr, new_xyz1, gidx,
      wh[3], bsrc[3], wh[4], bsrc[4], wh[5], bsrc[5],
      feats1, N, NP1, 96, 32);

  // ---- SA level 1 (1024 -> 256, feats: 96 -> 256ch) ----
  fps_kernel<<<B, 1024, 0, stream>>>(new_xyz1, fps_idx2, NP1, NP2);
  gather_xyz_kernel<<<(B * NP2 * 3 + 255) / 256, 256, 0, stream>>>(
      new_xyz1, fps_idx2, new_xyz2, NP1, NP2, B);

  ball_query_kernel<<<B * NP2 / 8, 256, 0, stream>>>(
      new_xyz1, new_xyz2, gidx, B, NP1, NP2, 0.5f * 0.5f, 16);
  sa_kernel<16, 96, 64, 64, 128><<<B * NP2 / 2, 64, 0, stream>>>(
      new_xyz1, feats1, new_xyz2, gidx,
      wh[6], bsrc[6], wh[7], bsrc[7], wh[8], bsrc[8],
      feats2, NP1, NP2, 256, 0);

  ball_query_kernel<<<B * NP2 / 8, 256, 0, stream>>>(
      new_xyz1, new_xyz2, gidx, B, NP1, NP2, 1.0f * 1.0f, 32);
  sa_kernel<32, 96, 64, 96, 128><<<B * NP2 / 2, 64, 0, stream>>>(
      new_xyz1, feats1, new_xyz2, gidx,
      wh[9], bsrc[9], wh[10], bsrc[10], wh[11], bsrc[11],
      feats2, NP1, NP2, 256, 128);

  // ---- FP level 1 (256 known -> 1024 unknown, [352,256,256]) ----
  interp_kernel<<<B * NP1 / 8, 256, 0, stream>>>(
      new_xyz1, new_xyz2, feats1, feats2, xin_fp1, B, NP1, NP2, 96, 256);
  fp_mlp_kernel<352, 256, 256><<<B * NP1 / (16 * 4), 128, 0, stream>>>(
      xin_fp1, wh[14], bsrc[14], wh[15], bsrc[15], fp1_out, B * NP1);

  // ---- FP level 0 (1024 known -> 8192 unknown, [256,128,128]) ----
  interp_kernel<<<B * N / 8, 256, 0, stream>>>(
      xyz, new_xyz1, nullptr, fp1_out, xin_fp0, B, N, NP1, 0, 256);
  fp_mlp_kernel<256, 128, 128><<<B * N / (16 * 4), 128, 0, stream>>>(
      xin_fp0, wh[12], bsrc[12], wh[13], bsrc[13], (float*)d_out, B * N);
}
